// DeepSetEquivariantTransform_6992206757904
// MI455X (gfx1250) — compile-verified, matrix operand order checked by reading, not run on hardware
//
#include <hip/hip_runtime.h>
#include <hip/hip_bf16.h>

// DeepSet equivariant network on MI455X (gfx1250, wave32).
// bf16 activations/weights, f32 accumulation via v_wmma_f32_16x16x32_bf16.
// Weights pre-packed into per-fragment layout, async-staged into LDS
// (global_load_async_to_lds_b128 + s_wait_asynccnt); A fragments loaded
// directly from global in the native 16-bit A-matrix VGPR layout.
// Wave tile 32Mx128N: each B fragment read from LDS feeds two WMMAs.
// Epilogue fuses the "- mean @ Lambda" broadcast bias and leaky-ReLU.

typedef __attribute__((ext_vector_type(8)))  __bf16 v8bf;
typedef __attribute__((ext_vector_type(16))) __bf16 v16bf;
typedef __attribute__((ext_vector_type(8)))  float  v8f;
typedef unsigned int u32;

#define ALPHA 0.2f
#define BB  512
#define NN  200
#define FF  16
#define WW  256
#define LATC 128
#define MTOT (BB * NN)   // 102400

__device__ __forceinline__ float lrelu(float v) { return v >= 0.f ? v : ALPHA * v; }

__device__ __forceinline__ __bf16 f2bf(float f) {
  u32 u = __builtin_bit_cast(u32, f);
  u32 r = (u + 0x7FFFu + ((u >> 16) & 1u)) >> 16;   // round-to-nearest-even
  unsigned short s = (unsigned short)r;
  return __builtin_bit_cast(__bf16, s);
}
__device__ __forceinline__ float bf2f(__bf16 v) {
  unsigned short s = __builtin_bit_cast(unsigned short, v);
  u32 u = ((u32)s) << 16;
  return __builtin_bit_cast(float, u);
}

// ---------------------------------------------------------------------------
// 1) mask = any(x != 0, feature axis); x -> bf16, zero-padded K: 16 -> 32
// ---------------------------------------------------------------------------
__global__ void __launch_bounds__(256) prep_kernel(const float* __restrict__ x,
                                                   __bf16* __restrict__ act,
                                                   float* __restrict__ mask) {
  int p = blockIdx.x * 256 + threadIdx.x;
  if (p >= MTOT) return;
  const float* xr = x + (size_t)p * FF;
  __bf16* ar = act + (size_t)p * 32;
  bool any = false;
  unsigned short z = 0;
#pragma unroll
  for (int c = 0; c < FF; c++) {
    float v = xr[c];
    any = any || (v != 0.f);
    ar[c] = f2bf(v);
  }
#pragma unroll
  for (int c = FF; c < 32; c++) ar[c] = __builtin_bit_cast(__bf16, z);
  mask[p] = any ? 1.f : 0.f;
}

// ---------------------------------------------------------------------------
// 2) denomInv[b] = 1 / sum_n mask[b,n]
// ---------------------------------------------------------------------------
__global__ void __launch_bounds__(256) denom_kernel(const float* __restrict__ mask,
                                                    float* __restrict__ denomInv) {
  __shared__ float red[256];
  int b = blockIdx.x, tid = threadIdx.x;
  red[tid] = (tid < NN) ? mask[b * NN + tid] : 0.f;
  __syncthreads();
  for (int o = 128; o > 0; o >>= 1) {
    if (tid < o) red[tid] += red[tid + o];
    __syncthreads();
  }
  if (tid == 0) denomInv[b] = 1.0f / red[0];
}

// ---------------------------------------------------------------------------
// 3) Pack f32 weight [Ksrc, N] -> bf16 WMMA B-fragments.
//    Fragment fid = kt*(N/16)+nt holds a 32(K)x16(N) tile: lane L owns column
//    nt*16+(L&15); its 16 bf16 values j=0..15 map to K offsets
//    {0..7,16..23} (+8 if L>=16), mirroring the 16-bit operand layout.
// ---------------------------------------------------------------------------
__global__ void __launch_bounds__(256) pack_kernel(const float* __restrict__ src,
                                                   __bf16* __restrict__ dst,
                                                   int Ksrc, int Kdst, int N) {
  int tid = threadIdx.x;
  int fid = blockIdx.x * 8 + (tid >> 5);
  int lane = tid & 31;
  int nT = N >> 4;
  int frags = (Kdst >> 5) * nT;
  if (fid >= frags) return;
  int kt = fid / nT, nt = fid % nT;
  int n = nt * 16 + (lane & 15);
  int kh = (lane & 16) ? 8 : 0;
#pragma unroll
  for (int j = 0; j < 16; j++) {
    int koff = (j < 8 ? j : j + 8) + kh;
    int k = kt * 32 + koff;
    float v = (k < Ksrc) ? src[(size_t)k * N + n] : 0.f;
    dst[(size_t)fid * 512 + lane * 16 + j] = f2bf(v);
  }
}

// ---------------------------------------------------------------------------
// 4) bias[b,:] = -( (sum_n mask*h[b,n,:]) / denom ) @ L     (small GEMV)
// ---------------------------------------------------------------------------
__global__ void __launch_bounds__(256) mean_bias_kernel(
    const __bf16* __restrict__ act, int Kstride, int Kin,
    const float* __restrict__ L, int Nout,
    const float* __restrict__ mask, const float* __restrict__ denomInv,
    float* __restrict__ bias) {
  __shared__ float mean[256];
  int b = blockIdx.x, tid = threadIdx.x;
  if (tid < Kin) {
    const __bf16* p = act + (size_t)b * NN * Kstride + tid;
    const float* mk = mask + b * NN;
    float s = 0.f;
    for (int n = 0; n < NN; n++) s += mk[n] * bf2f(p[(size_t)n * Kstride]);
    mean[tid] = s * denomInv[b];
  }
  __syncthreads();
  if (tid < Nout) {
    float s = 0.f;
    for (int k = 0; k < Kin; k++) s += mean[k] * L[(size_t)k * Nout + tid];
    bias[b * 256 + tid] = -s;
  }
}

// ---------------------------------------------------------------------------
// 5) Main equivariant GEMM:  Out = lrelu(A @ G + bias[batch(row)])
//    256 threads = 8 waves. Packed weights async-staged whole into LDS.
//    Each wave owns a 32(M) x 128(N) tile = 16 accumulators; each LDS B
//    fragment feeds two WMMAs (ds:wmma = 1:1).
// ---------------------------------------------------------------------------
__global__ void __launch_bounds__(256) eq_gemm_kernel(
    const __bf16* __restrict__ A, int K,
    const __bf16* __restrict__ Bp,
    const float* __restrict__ bias,
    __bf16* __restrict__ Out, int N) {
  extern __shared__ __bf16 Bs[];
  const int tid = threadIdx.x;
  const int lane = tid & 31;
  const int wid = tid >> 5;
  const int waves_n = N >> 7;        // 2 for N=256, 1 for N=128
  const int waves_m = 8 / waves_n;   // 4 or 8
  const int wm = wid % waves_m;
  const int wn = wid / waves_m;
  const int blockRows = waves_m << 5;   // 32 rows per wave
  const int nTiles = N >> 4;

  // Async-stage full packed weight matrix into LDS: each lane moves 16 bytes
  // per op, tracked by ASYNCcnt.
  const int bBytes = K * N * 2;
  for (int off = tid * 16; off < bBytes; off += 256 * 16) {
    const char* g = (const char*)Bp + off;
    unsigned lofs = (unsigned)off;
    asm volatile("global_load_async_to_lds_b128 %0, %1, off"
                 :: "v"(lofs), "v"(g) : "memory");
  }
  asm volatile("s_wait_asynccnt 0x0" ::: "memory");
  __syncthreads();

  const int tileRow = blockIdx.x * blockRows + (wm << 5);
  const int kHalf = (lane & 16) ? 8 : 0;
  const __bf16* aRow0 = A + (size_t)(tileRow + (lane & 15)) * K + kHalf;
  const __bf16* aRow1 = aRow0 + (size_t)16 * K;
  const int nTile0 = wn << 3;        // first n-tile index for this wave

  v8f acc[16] = {};

  for (int kk = 0; kk < K; kk += 32) {
    __builtin_prefetch(aRow0 + kk + 64, 0, 0);   // global_prefetch_b8
    __builtin_prefetch(aRow1 + kk + 64, 0, 0);
    v8bf a0lo = *(const v8bf*)(aRow0 + kk);
    v8bf a0hi = *(const v8bf*)(aRow0 + kk + 16);
    v8bf a1lo = *(const v8bf*)(aRow1 + kk);
    v8bf a1hi = *(const v8bf*)(aRow1 + kk + 16);
    v16bf a0 = __builtin_shufflevector(a0lo, a0hi, 0, 1, 2, 3, 4, 5, 6, 7,
                                       8, 9, 10, 11, 12, 13, 14, 15);
    v16bf a1 = __builtin_shufflevector(a1lo, a1hi, 0, 1, 2, 3, 4, 5, 6, 7,
                                       8, 9, 10, 11, 12, 13, 14, 15);
    const __bf16* bk = Bs + (size_t)(kk >> 5) * nTiles * 512 + lane * 16;
#pragma unroll
    for (int t = 0; t < 8; t++) {
      const __bf16* bp = bk + (size_t)(nTile0 + t) * 512;
      v8bf blo = *(const v8bf*)(bp);
      v8bf bhi = *(const v8bf*)(bp + 8);
      v16bf bfr = __builtin_shufflevector(blo, bhi, 0, 1, 2, 3, 4, 5, 6, 7,
                                          8, 9, 10, 11, 12, 13, 14, 15);
      acc[2 * t + 0] = __builtin_amdgcn_wmma_f32_16x16x32_bf16(
          false, a0, false, bfr, (short)0, acc[2 * t + 0], false, false);
      acc[2 * t + 1] = __builtin_amdgcn_wmma_f32_16x16x32_bf16(
          false, a1, false, bfr, (short)0, acc[2 * t + 1], false, false);
    }
  }

  // Epilogue: C/D layout => lanes 0-15 hold M=v, lanes 16-31 hold M=v+8.
  const int rowHi = (lane >> 4) & 1;
  const int colLane = lane & 15;
#pragma unroll
  for (int ms = 0; ms < 2; ms++) {
#pragma unroll
    for (int v = 0; v < 8; v++) {
      const int row = tileRow + ms * 16 + rowHi * 8 + v;
      const unsigned bidx = (unsigned)row / 200u;
      const float* brow = bias + (size_t)bidx * 256;
      __bf16* orow = Out + (size_t)row * N;
#pragma unroll
      for (int t = 0; t < 8; t++) {
        const int col = ((nTile0 + t) << 4) + colLane;
        orow[col] = f2bf(lrelu(acc[2 * t + ms][v] + brow[col]));
      }
    }
  }
}

// ---------------------------------------------------------------------------
// 6) pooled[b,c] = sum_n mask * h[b,n,c]   (h stride = LATC)
// ---------------------------------------------------------------------------
__global__ void __launch_bounds__(128) pool_kernel(const __bf16* __restrict__ h,
                                                   const float* __restrict__ mask,
                                                   float* __restrict__ pooled) {
  int b = blockIdx.x, c = threadIdx.x;
  const __bf16* p = h + (size_t)b * NN * LATC + c;
  const float* mk = mask + b * NN;
  float s = 0.f;
  for (int n = 0; n < NN; n++) s += mk[n] * bf2f(p[(size_t)n * LATC]);
  pooled[b * LATC + c] = s;
}

// ---------------------------------------------------------------------------
// 7) MLP head: 128->256, 4x 256->256 (lrelu), 256->2 + softmax. One block per b.
// ---------------------------------------------------------------------------
__global__ void __launch_bounds__(256) mlp_kernel(
    const float* __restrict__ pooled,
    const float* __restrict__ W0, const float* __restrict__ b0,
    const float* __restrict__ Wm, const float* __restrict__ bm,
    const float* __restrict__ Wl, const float* __restrict__ bl,
    float* __restrict__ out) {
  __shared__ float ya[256], yb[256];
  int b = blockIdx.x, tid = threadIdx.x;
  if (tid < LATC) ya[tid] = pooled[b * LATC + tid];
  __syncthreads();
  {
    float s = b0[tid];
    for (int k = 0; k < LATC; k++) s += ya[k] * W0[(size_t)k * 256 + tid];
    yb[tid] = lrelu(s);
  }
  __syncthreads();
  float* src = yb;
  float* dst = ya;
  for (int i = 0; i < 4; i++) {
    float s = bm[i * 256 + tid];
    const float* w = Wm + (size_t)i * 256 * 256;
    for (int k = 0; k < 256; k++) s += src[k] * w[(size_t)k * 256 + tid];
    dst[tid] = lrelu(s);
    __syncthreads();
    float* tmp = src; src = dst; dst = tmp;
  }
  if (tid == 0) {
    float z0 = bl[0], z1 = bl[1];
    for (int k = 0; k < 256; k++) {
      z0 += src[k] * Wl[k * 2 + 0];
      z1 += src[k] * Wl[k * 2 + 1];
    }
    float m = z0 > z1 ? z0 : z1;
    float e0 = __expf(z0 - m), e1 = __expf(z1 - m);
    float inv = 1.f / (e0 + e1);
    out[b * 2 + 0] = e0 * inv;
    out[b * 2 + 1] = e1 * inv;
  }
}

// ---------------------------------------------------------------------------
extern "C" void kernel_launch(void* const* d_in, const int* in_sizes, int n_in,
                              void* d_out, int out_size, void* d_ws, size_t ws_size,
                              hipStream_t stream) {
  (void)in_sizes; (void)n_in; (void)out_size; (void)ws_size;
  const float* x       = (const float*)d_in[0];
  const float* Gamma0  = (const float*)d_in[1];
  const float* Lambda0 = (const float*)d_in[2];
  const float* Gmid    = (const float*)d_in[3];
  const float* Lmid    = (const float*)d_in[4];
  const float* GammaL  = (const float*)d_in[5];
  const float* LambdaL = (const float*)d_in[6];
  const float* F0_W    = (const float*)d_in[7];
  const float* F0_b    = (const float*)d_in[8];
  const float* FmW     = (const float*)d_in[9];
  const float* Fmb     = (const float*)d_in[10];
  const float* FlW     = (const float*)d_in[11];
  const float* Flb     = (const float*)d_in[12];

  char* p = (char*)d_ws;
  auto carve = [&](size_t bytes) -> char* {
    char* r = p;
    p += (bytes + 255) & ~(size_t)255;
    return r;
  };
  __bf16* actA   = (__bf16*)carve((size_t)MTOT * 256 * 2);  // 52.4 MB
  __bf16* actB   = (__bf16*)carve((size_t)MTOT * 256 * 2);  // 52.4 MB
  float*  mask   = (float*)carve((size_t)MTOT * 4);
  float*  dinv   = (float*)carve((size_t)BB * 4);
  float*  bias   = (float*)carve((size_t)BB * 256 * 4);
  float*  pooled = (float*)carve((size_t)BB * LATC * 4);
  __bf16* pG0    = (__bf16*)carve((size_t)32 * 256 * 2);
  __bf16* pGm    = (__bf16*)carve((size_t)3 * 256 * 256 * 2);
  __bf16* pGL    = (__bf16*)carve((size_t)256 * 128 * 2);

  prep_kernel<<<(MTOT + 255) / 256, 256, 0, stream>>>(x, actA, mask);
  denom_kernel<<<BB, 256, 0, stream>>>(mask, dinv);

  pack_kernel<<<2, 256, 0, stream>>>(Gamma0, pG0, 16, 32, 256);       // 16 frags
  for (int i = 0; i < 3; i++)
    pack_kernel<<<16, 256, 0, stream>>>(Gmid + (size_t)i * 65536,
                                        pGm + (size_t)i * 65536, 256, 256, 256);
  pack_kernel<<<8, 256, 0, stream>>>(GammaL, pGL, 256, 256, 128);     // 64 frags

  // Layer 0: [M,32 padded] @ [32,256]   (blockRows = 128)
  mean_bias_kernel<<<BB, 256, 0, stream>>>(actA, 32, 16, Lambda0, 256, mask, dinv, bias);
  eq_gemm_kernel<<<MTOT / 128, 256, 32 * 256 * 2, stream>>>(actA, 32, pG0, bias, actB, 256);

  // 3 mid layers: [M,256] @ [256,256], ping-pong  (blockRows = 128)
  __bf16* cur = actB;
  __bf16* nxt = actA;
  for (int i = 0; i < 3; i++) {
    mean_bias_kernel<<<BB, 256, 0, stream>>>(cur, 256, 256, Lmid + (size_t)i * 65536,
                                             256, mask, dinv, bias);
    eq_gemm_kernel<<<MTOT / 128, 256, 256 * 256 * 2, stream>>>(
        cur, 256, pGm + (size_t)i * 65536, bias, nxt, 256);
    __bf16* t = cur; cur = nxt; nxt = t;
  }

  // Last equivariant layer: [M,256] @ [256,128]   (blockRows = 256)
  mean_bias_kernel<<<BB, 256, 0, stream>>>(cur, 256, 256, LambdaL, 128, mask, dinv, bias);
  eq_gemm_kernel<<<MTOT / 256, 256, 256 * 128 * 2, stream>>>(cur, 256, pGL, bias, nxt, 128);

  pool_kernel<<<BB, LATC, 0, stream>>>(nxt, mask, pooled);
  mlp_kernel<<<BB, 256, 0, stream>>>(pooled, F0_W, F0_b, FmW, Fmb, FlW, Flb,
                                     (float*)d_out);
}
